// GraphUNet_5360119185581
// MI455X (gfx1250) — matrix operand order repaired and checked by reading, not compiled, not run on hardware
//
#include <hip/hip_runtime.h>
#include <hip/hip_bf16.h>

typedef __bf16 bf16;
typedef unsigned int uint;
typedef __attribute__((ext_vector_type(16))) __bf16 v16bf;
typedef __attribute__((ext_vector_type(8)))  float  v8f;
typedef __attribute__((ext_vector_type(8)))  uint   v8u;

// ---------------- shared-memory layout (bf16 elements) -----------------------
// One packed block; deliberate small K-overrun reads stay inside the block and
// are multiplied by zero padding in the packed operands.
constexpr int XIN = 0;                 // [32 rows][16 s][2]   (29 used)
constexpr int C10 = XIN + 1024;        // [32][16][8]  X0 cols0..3 | u10 cols4..7
constexpr int C9  = C10 + 4096;        // [16][16][16] X1 cols0..7 | u9 cols8..15
constexpr int C8  = C9  + 4096;        // [7][16][32]  X2 cols0..15 | u8 16..31
constexpr int C7  = C8  + 3584;        // [4][16][64]  X3 | u7
constexpr int C6  = C7  + 4096;        // [2][16][128] X4 | u6
constexpr int W1  = C6  + 4096;        // scratch 4096
constexpr int W2  = W1  + 4096;        // scratch 2048
constexpr int SLK = W2  + 2048;        // slack for overrun reads (stays zero)
constexpr int SM_TOTAL = SLK + 2048;   // 29184 bf16 = 58368 B

// ---------------- parameter structs ------------------------------------------
struct MainP {
  const float* gb[10];
  const float* pb[5];
  const float* ub[5];
  const float *fc1b, *fc2b;
  const float *pw3, *pw4, *pw5, *uw6, *uw7, *uw8;     // tiny node mats (VALU)
  const float *L2f, *L3f, *L4f;                        // tiny Laplacians (VALU)
  const bf16 *L0p, *L1p, *p1p, *p2p, *u9p, *u10p;      // A-fragment packs
  const bf16* Wp[12];                                  // B-fragment packs
};

struct PrepP {
  const float* A[5];
  const float* gw[10];
  const float *fc1w, *fc2w, *pw1, *pw2, *uw9, *uw10;
  float *L2f, *L3f, *L4f;
  bf16 *L0p, *L1p, *p1p, *p2p, *u9p, *u10p;
  bf16* Wp[12];
};

__device__ __forceinline__ v8f wmma_bf16(v16bf a, v16bf b, v8f c) {
  return __builtin_amdgcn_wmma_f32_16x16x32_bf16(false, a, false, b,
                                                 (short)0, c, false, false);
}

// ---------------- WMMA feature GEMM ------------------------------------------
// rows=(node,sample), K=Fin, N=Fout. src layout (n,s,f) at (n*16+s)*stride+f.
// Wp is B-fragment packed [KT][NT][32 lanes][16 bf16], zero-padded.
template<int NN, int FIN, int FOUT, bool RELU>
__device__ __forceinline__ void feat_gemm(
    bf16* dst, int dstStride, int dstOff,
    const bf16* src, int srcStride,
    const bf16* Wp, const float* bias)
{
  constexpr int KT = (FIN + 31) / 32;
  constexpr int NT = (FOUT + 15) / 16;
  const int lane = threadIdx.x;
  const int h  = lane >> 4;
  const int ln = lane & 15;
  for (int n = 0; n < NN; ++n) {
    const bf16* srcRow = src + (n * 16 + ln) * srcStride;
#pragma unroll
    for (int nt = 0; nt < NT; ++nt) {
      v8f c = {};
#pragma unroll
      for (int kt = 0; kt < KT; ++kt) {
        v8u au;
#pragma unroll
        for (int v = 0; v < 8; ++v) {
          const int ka = kt * 32 + ((v < 4) ? (8 * h + 2 * v)
                                            : (16 + 8 * h + 2 * (v - 4)));
          au[v] = *(const uint*)(srcRow + ka);          // paired ds_load_b32
        }
        const v16bf a = __builtin_bit_cast(v16bf, au);
        const v16bf b = *(const v16bf*)(Wp + ((kt * NT + nt) * 32 + lane) * 16);
        c = wmma_bf16(a, b, c);
      }
      const int col = nt * 16 + ln;
      if ((FOUT % 16 == 0) || (col < FOUT)) {
        const float bs = bias[col];
#pragma unroll
        for (int r = 0; r < 8; ++r) {
          float vv = c[r] + bs;
          if (RELU) vv = fmaxf(vv, 0.0f);
          dst[(n * 16 + r + 8 * h) * dstStride + dstOff + col] = (bf16)vv;
        }
      }
    }
  }
}

// ---------------- WMMA node-mix GEMM -----------------------------------------
// Y[m,(s,f)] = sum_k Mt[m,k] * X[k,(s,f)]; Mp is A-fragment packed, zero-padded.
// Stores are unconditional: dst buffers sized for the padded row range.
template<int NOUT, int NIN, int F, bool BIAS>
__device__ __forceinline__ void node_gemm(
    bf16* dst, int dstStride, int dstOff,
    const bf16* src, int srcStride,
    const bf16* Mp, const float* bias)
{
  constexpr int MT = (NOUT + 15) / 16;
  constexpr int KT = (NIN  + 31) / 32;
  const int lane = threadIdx.x;
  const int h  = lane >> 4;
  const int ln = lane & 15;
  const int rowStep = 16 * srcStride;
#pragma unroll
  for (int mt = 0; mt < MT; ++mt) {
    float bv[8];
    if constexpr (BIAS) {
#pragma unroll
      for (int r = 0; r < 8; ++r) {
        const int m = mt * 16 + r + 8 * h;
        bv[r] = (m < NOUT) ? bias[m] : 0.0f;
      }
    }
#pragma unroll
    for (int ct = 0; ct < F; ++ct) {
      const int cg = ct * 16 + ln;       // global column = s*F + f
      const int s  = cg / F;             // F is a power of two -> shifts
      const int f  = cg % F;
      const bf16* col0 = src + s * srcStride + f;
      v8f c = {};
#pragma unroll
      for (int kt = 0; kt < KT; ++kt) {
        const v16bf a = *(const v16bf*)(Mp + ((mt * KT + kt) * 32 + lane) * 16);
        v16bf b;
#pragma unroll
        for (int v = 0; v < 8; ++v) {
          const int kb = kt * 32 + 16 * h + 2 * v;
          b[2 * v]     = col0[kb * rowStep];           // unconditional ds_load
          b[2 * v + 1] = col0[(kb + 1) * rowStep];
        }
        c = wmma_bf16(a, b, c);
      }
#pragma unroll
      for (int r = 0; r < 8; ++r) {
        const int m = mt * 16 + r + 8 * h;
        float vv;
        if constexpr (BIAS) vv = c[r] + bv[r];
        else                vv = c[r];                 // no +0.0 add emitted
        dst[(m * 16 + s) * dstStride + dstOff + f] = (bf16)vv;
      }
    }
  }
}

// -------- f32 VALU node-mix for tiny node dims (WMMA <2% utilized there) -----
template<int NOUT, int NIN, int F, bool BIAS>
__device__ __forceinline__ void node_valu(
    bf16* dst, int dstStride, int dstOff,
    const bf16* src, int srcStride,
    const float* M, const float* bias)
{
  constexpr int TOTAL = NOUT * 16 * F;
  for (int i = threadIdx.x; i < TOTAL; i += 32) {
    const int m   = i / (16 * F);
    const int rem = i - m * 16 * F;
    const int s   = rem / F;
    const int f   = rem - s * F;
    float acc = BIAS ? bias[m] : 0.0f;
#pragma unroll
    for (int k = 0; k < NIN; ++k)
      acc += M[m * NIN + k] * (float)src[(k * 16 + s) * srcStride + f];
    dst[(m * 16 + s) * dstStride + dstOff + f] = (bf16)acc;
  }
}

#define SYNC() __syncthreads()

// ---------------- fused GraphUNet: one wave = 16 graphs ----------------------
__global__ __launch_bounds__(32) void graphunet_fused(
    const float* __restrict__ X, float* __restrict__ out, int B, MainP pp)
{
  __shared__ __align__(16) bf16 SM[SM_TOTAL];

  // zero-fill LDS once (uint4 stores) so every overrun read is finite zero
  {
    uint4* p = (uint4*)SM;
    const uint4 z = make_uint4(0u, 0u, 0u, 0u);
    for (int i = threadIdx.x; i < SM_TOTAL / 8; i += 32) p[i] = z;
  }
  SYNC();

  const int sBase = blockIdx.x * 16;

  // stage input tile (f32 -> bf16), node-major [n][s][2]
  for (int i = threadIdx.x; i < 29 * 16 * 2; i += 32) {
    const int n = i >> 5, rem = i & 31, s = rem >> 1, f = rem & 1;
    float v = 0.0f;
    const int b = sBase + s;
    if (b < B) v = X[((size_t)b * 29 + n) * 2 + f];
    SM[XIN + (n * 16 + s) * 2 + f] = (bf16)v;
  }
  SYNC();

  // ---------------- encoder ----------------
  node_gemm<29,29,2,false>(SM+W1, 2, 0, SM+XIN, 2, pp.L0p, nullptr);        SYNC();
  feat_gemm<29,2,4,true>  (SM+C10, 8, 0, SM+W1, 2, pp.Wp[0], pp.gb[0]);     SYNC(); // X0
  node_gemm<15,29,4,true> (SM+W2, 4, 0, SM+C10, 8, pp.p1p, pp.pb[0]);       SYNC();
  node_gemm<15,15,4,false>(SM+W1, 4, 0, SM+W2, 4, pp.L1p, nullptr);         SYNC();
  feat_gemm<15,4,8,true>  (SM+C9, 16, 0, SM+W1, 4, pp.Wp[1], pp.gb[1]);     SYNC(); // X1
  node_gemm<7,15,8,true>  (SM+W2, 8, 0, SM+C9, 16, pp.p2p, pp.pb[1]);       SYNC();
  node_valu<7,7,8,false>  (SM+W1, 8, 0, SM+W2, 8, pp.L2f, nullptr);         SYNC();
  feat_gemm<7,8,16,true>  (SM+C8, 32, 0, SM+W1, 8, pp.Wp[2], pp.gb[2]);     SYNC(); // X2
  node_valu<4,7,16,true>  (SM+W2, 16, 0, SM+C8, 32, pp.pw3, pp.pb[2]);      SYNC();
  node_valu<4,4,16,false> (SM+W1, 16, 0, SM+W2, 16, pp.L3f, nullptr);       SYNC();
  feat_gemm<4,16,32,true> (SM+C7, 64, 0, SM+W1, 16, pp.Wp[3], pp.gb[3]);    SYNC(); // X3
  node_valu<2,4,32,true>  (SM+W2, 32, 0, SM+C7, 64, pp.pw4, pp.pb[3]);      SYNC();
  node_valu<2,2,32,false> (SM+W1, 32, 0, SM+W2, 32, pp.L4f, nullptr);       SYNC();
  feat_gemm<2,32,64,true> (SM+C6, 128, 0, SM+W1, 32, pp.Wp[4], pp.gb[4]);   SYNC(); // X4
  // ---------------- bottleneck ----------------
  node_valu<1,2,64,true>  (SM+W2, 64, 0, SM+C6, 128, pp.pw5, pp.pb[4]);     SYNC(); // X5
  feat_gemm<1,64,20,true> (SM+W1, 20, 0, SM+W2, 64, pp.Wp[10], pp.fc1b);    SYNC();
  feat_gemm<1,20,64,true> (SM+W2, 64, 0, SM+W1, 20, pp.Wp[11], pp.fc2b);    SYNC();
  // ---------------- decoder ----------------
  node_valu<2,1,64,true>  (SM+C6, 128, 64, SM+W2, 64, pp.uw6, pp.ub[0]);    SYNC();
  node_valu<2,2,128,false>(SM+W1, 128, 0, SM+C6, 128, pp.L4f, nullptr);     SYNC();
  feat_gemm<2,128,32,true>(SM+W2, 32, 0, SM+W1, 128, pp.Wp[5], pp.gb[5]);   SYNC(); // X6
  node_valu<4,2,32,true>  (SM+C7, 64, 32, SM+W2, 32, pp.uw7, pp.ub[1]);     SYNC();
  node_valu<4,4,64,false> (SM+W1, 64, 0, SM+C7, 64, pp.L3f, nullptr);       SYNC();
  feat_gemm<4,64,16,true> (SM+W2, 16, 0, SM+W1, 64, pp.Wp[6], pp.gb[6]);    SYNC(); // X7
  node_valu<7,4,16,true>  (SM+C8, 32, 16, SM+W2, 16, pp.uw8, pp.ub[2]);     SYNC();
  node_valu<7,7,32,false> (SM+W1, 32, 0, SM+C8, 32, pp.L2f, nullptr);       SYNC();
  feat_gemm<7,32,8,true>  (SM+W2, 8, 0, SM+W1, 32, pp.Wp[7], pp.gb[7]);     SYNC(); // X8
  node_gemm<15,7,8,true>  (SM+C9, 16, 8, SM+W2, 8, pp.u9p, pp.ub[3]);       SYNC();
  node_gemm<15,15,16,false>(SM+W1, 16, 0, SM+C9, 16, pp.L1p, nullptr);      SYNC();
  feat_gemm<15,16,4,true> (SM+W2, 4, 0, SM+W1, 16, pp.Wp[8], pp.gb[8]);     SYNC(); // X9
  node_gemm<29,15,4,true> (SM+C10, 8, 4, SM+W2, 4, pp.u10p, pp.ub[4]);      SYNC();
  node_gemm<29,29,8,false>(SM+W1, 8, 0, SM+C10, 8, pp.L0p, nullptr);        SYNC();
  feat_gemm<29,8,3,false> (SM+W2, 3, 0, SM+W1, 8, pp.Wp[9], pp.gb[9]);      SYNC(); // X10

  // store output tile (bf16 -> f32), [B][29][3]
  for (int i = threadIdx.x; i < 29 * 16 * 3; i += 32) {
    const int n = i / 48, rem = i % 48, s = rem / 3, f = rem % 3;
    const int b = sBase + s;
    if (b < B) out[((size_t)b * 29 + n) * 3 + f] = (float)SM[W2 + (n * 16 + s) * 3 + f];
  }
}

// ---------------- prepass: fragment-packed operands --------------------------
// A-fragment pack: [MT][KT][32 lanes][16 bf16], contiguous k = 16h + j within each lane half.
__device__ void packA(bf16* dst, const float* M, int Nout, int Nin, int MT, int KT)
{
  const int total = MT * KT * 512;
  for (int t = threadIdx.x; t < total; t += blockDim.x) {
    const int tile = t >> 9, r = t & 511, lane = r >> 4, j = r & 15;
    const int h = lane >> 4, ln = lane & 15;
    const int mt = tile / KT, kt = tile % KT;
    const int v = j >> 1, par = j & 1;
    const int ka = kt * 32 + 16 * h + 2 * v + par;
    const int m = mt * 16 + ln;
    const float val = (m < Nout && ka < Nin) ? M[m * Nin + ka] : 0.0f;
    dst[t] = (bf16)val;
  }
}

// B-fragment pack: [KT][NT][32 lanes][16 bf16]; source W is [Fout][Fin] (=gw).
__device__ void packB(bf16* dst, const float* W, int Fin, int Fout)
{
  const int KT = (Fin + 31) / 32, NT = (Fout + 15) / 16;
  const int total = KT * NT * 512;
  for (int t = threadIdx.x; t < total; t += blockDim.x) {
    const int tile = t >> 9, r = t & 511, lane = r >> 4, j = r & 15;
    const int h = lane >> 4, ln = lane & 15;
    const int kt = tile / NT, nt = tile % NT;
    const int v = j >> 1, par = j & 1;
    const int kb = kt * 32 + ((v < 4) ? (8 * h + 2 * v)
                                      : (16 + 8 * h + 2 * (v - 4))) + par;
    const int col = nt * 16 + ln;
    const float val = (col < Fout && kb < Fin) ? W[col * Fin + kb] : 0.0f;
    dst[t] = (bf16)val;
  }
}

__global__ __launch_bounds__(256) void graphunet_prep(PrepP p)
{
  __shared__ float dD[32];
  __shared__ float Lbuf[841];
  const int Ns[5] = {29, 15, 7, 4, 2};
  for (int m = 0; m < 5; ++m) {
    const int N = Ns[m];
    if ((int)threadIdx.x < N) {
      float s = 1e-5f;
      for (int i = 0; i < N; ++i) s += p.A[m][i * N + threadIdx.x];
      dD[threadIdx.x] = 1.0f / sqrtf(s);
    }
    __syncthreads();
    for (int t = threadIdx.x; t < N * N; t += blockDim.x)
      Lbuf[t] = dD[t / N] * p.A[m][t] * dD[t % N];
    __syncthreads();
    if (m == 0)      packA(p.L0p, Lbuf, 29, 29, 2, 1);
    else if (m == 1) packA(p.L1p, Lbuf, 15, 15, 1, 1);
    else {
      float* dst = (m == 2) ? p.L2f : (m == 3) ? p.L3f : p.L4f;
      for (int t = threadIdx.x; t < N * N; t += blockDim.x) dst[t] = Lbuf[t];
    }
    __syncthreads();
  }
  packA(p.p1p,  p.pw1,  15, 29, 1, 1);
  packA(p.p2p,  p.pw2,   7, 15, 1, 1);
  packA(p.u9p,  p.uw9,  15,  7, 1, 1);
  packA(p.u10p, p.uw10, 29, 15, 2, 1);
  const int Fi[12] = {2, 4, 8, 16, 32, 128, 64, 32, 16, 8, 64, 20};
  const int Fo[12] = {4, 8, 16, 32, 64, 32, 16, 8, 4, 3, 20, 64};
  for (int w = 0; w < 12; ++w) {
    const float* src = (w < 10) ? p.gw[w] : (w == 10 ? p.fc1w : p.fc2w);
    packB(p.Wp[w], src, Fi[w], Fo[w]);
  }
}

// ---------------- host side --------------------------------------------------
extern "C" void kernel_launch(void* const* d_in, const int* in_sizes, int n_in,
                              void* d_out, int out_size, void* d_ws, size_t ws_size,
                              hipStream_t stream)
{
  const int B = in_sizes[0] / (29 * 2);

  char* w = (char*)d_ws;
  size_t off = 0;
  auto carveF = [&](size_t n) -> float* { float* q = (float*)(w + off); off += n * 4; return q; };
  float* L2f = carveF(49);
  float* L3f = carveF(16);
  float* L4f = carveF(4);
  off = (off + 63) & ~(size_t)63;                      // align packs to 64 B
  auto carveB = [&](size_t n) -> bf16* { bf16* q = (bf16*)(w + off); off += n * 2; return q; };
  bf16* L0p  = carveB(1024);                           // MT=2, KT=1
  bf16* L1p  = carveB(512);
  bf16* p1p  = carveB(512);
  bf16* p2p  = carveB(512);
  bf16* u9p  = carveB(512);
  bf16* u10p = carveB(1024);
  const int WpN[12] = {512, 512, 512, 1024, 2048, 4096, 1024, 512, 512, 512, 2048, 2048};
  bf16* Wp[12];
  for (int i = 0; i < 12; ++i) Wp[i] = carveB(WpN[i]);
  if (off > ws_size) return;

  PrepP pr;
  for (int i = 0; i < 5; ++i)  pr.A[i]  = (const float*)d_in[1 + i];
  for (int i = 0; i < 10; ++i) pr.gw[i] = (const float*)d_in[7 + 2 * i];
  pr.fc1w = (const float*)d_in[47]; pr.fc2w = (const float*)d_in[49];
  pr.pw1  = (const float*)d_in[27]; pr.pw2  = (const float*)d_in[29];
  pr.uw9  = (const float*)d_in[43]; pr.uw10 = (const float*)d_in[45];
  pr.L2f = L2f; pr.L3f = L3f; pr.L4f = L4f;
  pr.L0p = L0p; pr.L1p = L1p; pr.p1p = p1p; pr.p2p = p2p; pr.u9p = u9p; pr.u10p = u10p;
  for (int i = 0; i < 12; ++i) pr.Wp[i] = Wp[i];

  MainP mp;
  for (int i = 0; i < 10; ++i) mp.gb[i] = (const float*)d_in[8 + 2 * i];
  for (int i = 0; i < 5; ++i)  mp.pb[i] = (const float*)d_in[28 + 2 * i];
  for (int i = 0; i < 5; ++i)  mp.ub[i] = (const float*)d_in[38 + 2 * i];
  mp.fc1b = (const float*)d_in[48]; mp.fc2b = (const float*)d_in[50];
  mp.pw3 = (const float*)d_in[31]; mp.pw4 = (const float*)d_in[33]; mp.pw5 = (const float*)d_in[35];
  mp.uw6 = (const float*)d_in[37]; mp.uw7 = (const float*)d_in[39]; mp.uw8 = (const float*)d_in[41];
  mp.L2f = L2f; mp.L3f = L3f; mp.L4f = L4f;
  mp.L0p = L0p; mp.L1p = L1p; mp.p1p = p1p; mp.p2p = p2p; mp.u9p = u9p; mp.u10p = u10p;
  for (int i = 0; i < 12; ++i) mp.Wp[i] = Wp[i];

  graphunet_prep<<<1, 256, 0, stream>>>(pr);
  const int tiles = (B + 15) / 16;
  graphunet_fused<<<tiles, 32, 0, stream>>>((const float*)d_in[0], (float*)d_out, B, mp);
}
